// Transformer_15161234555450
// MI455X (gfx1250) — matrix operand order repaired and checked by reading, not compiled, not run on hardware
//
#include <hip/hip_runtime.h>
#include <hip/hip_bf16.h>

// Problem constants (match reference)
#define B_   4
#define S_   512
#define E_   512
#define H_   8
#define F_   2048
#define L_   4
#define V_   32000
#define HD_  64
#define EPS_ 1e-5f
#define NEG_ -1e9f

typedef __attribute__((ext_vector_type(16))) __bf16 v16bf;
typedef __attribute__((ext_vector_type(8)))  __bf16 v8bf;
typedef __attribute__((ext_vector_type(8)))  float  v8f;

// Native fp32 -> bf16 (pairs fuse to v_cvt_pk_bf16_f32)
static __device__ __forceinline__ __bf16 f2bf(float f) { return (__bf16)f; }

static __device__ __forceinline__ v8f wmma_bf16(v16bf a, v16bf b, v8f c) {
  return __builtin_amdgcn_wmma_f32_16x16x32_bf16(false, a, false, b, (short)0, c,
                                                 false, false);
}

static __device__ __forceinline__ v16bf cat8(v8bf lo, v8bf hi) {
  return __builtin_shufflevector(lo, hi, 0, 1, 2, 3, 4, 5, 6, 7,
                                 8, 9, 10, 11, 12, 13, 14, 15);
}

static __device__ __forceinline__ v8bf pack8(float4 f0, float4 f1) {
  v8bf p;
  p[0] = f2bf(f0.x); p[1] = f2bf(f0.y); p[2] = f2bf(f0.z); p[3] = f2bf(f0.w);
  p[4] = f2bf(f1.x); p[5] = f2bf(f1.y); p[6] = f2bf(f1.z); p[7] = f2bf(f1.w);
  return p;
}

// ---------------------------------------------------------------------------
// Token embedding + sinusoidal positional encoding
// ---------------------------------------------------------------------------
__global__ void embed_pe(const int* __restrict__ ids,
                         const float* __restrict__ emb,
                         float* __restrict__ out) {
  int idx = blockIdx.x * blockDim.x + threadIdx.x;   // over B*S*E
  int e = idx % E_;
  int t = idx / E_;        // b*S + s
  int s = t % S_;
  int id = ids[t];
  float expo = (float)(2 * (e >> 1)) * (1.0f / (float)E_);
  float inv  = __expf(-expo * 9.210340371976184f);   // ln(10000)
  float ang  = (float)s * inv;
  float pe   = (e & 1) ? __cosf(ang) : __sinf(ang);
  out[idx] = emb[(size_t)id * E_ + e] + pe;
}

// ---------------------------------------------------------------------------
// Tiled GEMM: C[M,N] = act(A[M,K] @ W[K,N] + bias[N]), N/K compile-time so
// strided W loads fold into immediate-offset clauses (no 64-bit add chains).
// 256 threads = 8 waves. Macro tile BM=64 x BN=128, BK=32.
// Stage A (64x32) and B^T (128x32) in LDS as bf16 (convert once, coalesced
// loads). Fragments are two contiguous ds_load_b128 per lane (ISA 7.12.2).
// Each wave owns a 16x64 strip: 4 WMMAs per BK step, A fragment reused.
// ---------------------------------------------------------------------------
template <int N, int K, bool RELU, bool BIAS>
__global__ __launch_bounds__(256) void gemm_bf16_wmma(
    const float* __restrict__ A, const float* __restrict__ W,
    const float* __restrict__ bias, float* __restrict__ C) {
  __shared__ __bf16 sA[64 * 32];     // [row][k]   4 KB
  __shared__ __bf16 sBT[128 * 32];   // [n][k]     8 KB

  const int tid   = threadIdx.x;
  const int lane  = tid & 31;
  const int w     = tid >> 5;        // wave 0..7
  const int wm    = w & 3;           // 4 row tiles
  const int wn    = w >> 2;          // 2 col strips of 64
  const int mlane = lane & 15;
  const int kh    = lane >> 4;
  const int row0  = blockIdx.y * 64;
  const int col0  = blockIdx.x * 128;

  // staging maps
  const int ar   = tid >> 2;         // A: row 0..63
  const int aseg = tid & 3;          // A: 8-float segment within the 32-k row
  const int bn   = tid & 127;        // B: n column 0..127
  const int bkh  = tid >> 7;         // B: k half (0 or 1) -> 16 k's

  const float* aSrc = &A[(size_t)(row0 + ar) * K + aseg * 8];
  const float* wSrc = &W[(size_t)(bkh * 16) * N + col0 + bn];

  const v8f vzero = {0.f, 0.f, 0.f, 0.f, 0.f, 0.f, 0.f, 0.f};
  v8f acc[4];
#pragma unroll
  for (int t = 0; t < 4; ++t) acc[t] = vzero;

  for (int k0 = 0; k0 < K; k0 += 32) {
    __syncthreads();   // previous compute done before overwriting tiles
    // ---- stage A: each thread 8 consecutive floats of one row ----
    {
      const float4 f0 = *reinterpret_cast<const float4*>(aSrc + k0);
      const float4 f1 = *reinterpret_cast<const float4*>(aSrc + k0 + 4);
      *reinterpret_cast<v8bf*>(&sA[ar * 32 + aseg * 8]) = pack8(f0, f1);
      __builtin_prefetch(aSrc + k0 + 32, 0, 1);   // speculative: OOB dropped
    }
    // ---- stage B transposed: thread = one column, 16 k's (coalesced) ----
    {
      const float* ws0 = wSrc + (size_t)k0 * N;
      v8bf p0, p1;
#pragma unroll
      for (int j = 0; j < 8; ++j) p0[j] = f2bf(ws0[(size_t)j * N]);
#pragma unroll
      for (int j = 0; j < 8; ++j) p1[j] = f2bf(ws0[(size_t)(8 + j) * N]);
      *reinterpret_cast<v8bf*>(&sBT[bn * 32 + bkh * 16]) = p0;
      *reinterpret_cast<v8bf*>(&sBT[bn * 32 + bkh * 16 + 8]) = p1;
      __builtin_prefetch(ws0 + (size_t)32 * N, 0, 1);
    }
    __syncthreads();

    // ---- compute: 4 WMMAs per wave, A fragment reused ----
    const int am = wm * 16 + mlane;
    const v8bf alo = *reinterpret_cast<const v8bf*>(&sA[am * 32 + 16 * kh]);
    const v8bf ahi = *reinterpret_cast<const v8bf*>(&sA[am * 32 + 16 * kh + 8]);
    const v16bf a = cat8(alo, ahi);
#pragma unroll
    for (int t = 0; t < 4; ++t) {
      const int n = wn * 64 + t * 16 + mlane;
      const v8bf blo = *reinterpret_cast<const v8bf*>(&sBT[n * 32 + 16 * kh]);
      const v8bf bhi = *reinterpret_cast<const v8bf*>(&sBT[n * 32 + 16 * kh + 8]);
      acc[t] = wmma_bf16(a, cat8(blo, bhi), acc[t]);
    }
  }

  // ---- epilogue ----
#pragma unroll
  for (int t = 0; t < 4; ++t) {
    const int col = col0 + wn * 64 + t * 16 + mlane;
    float bv = BIAS ? bias[col] : 0.f;
#pragma unroll
    for (int i = 0; i < 8; ++i) {
      int m = row0 + wm * 16 + i + 8 * kh;
      float v = acc[t][i] + bv;
      if (RELU) v = fmaxf(v, 0.f);
      C[(size_t)m * N + col] = v;
    }
  }
}

// ---------------------------------------------------------------------------
// Multi-head attention, one block (128 thr = 4 waves) per (b, h, 16-q tile).
// Phase 1: stage K panel (bf16, 256-key halves) in LDS; scores via WMMA
//          (Q fragments preloaded once per wave, reused across key tiles).
// Phase 2: softmax over the 16x512 score rows (4 rows/wave), normalized
//          probabilities written to LDS as bf16.
// Phase 3: stage V transposed (sVT[d][key]) in the same LDS buffer; P @ V
//          via WMMA, wave w produces output columns [w*16, w*16+16).
// All WMMA fragments are contiguous 16-byte LDS reads. Masking is
// compile-time selected; the pad byte is loaded once per key tile per lane.
// ---------------------------------------------------------------------------
template <bool CAUSAL, bool PAD>
__global__ __launch_bounds__(128) void attn_wmma(
    const float* __restrict__ Q, const float* __restrict__ Kt,
    const float* __restrict__ Vt, float* __restrict__ O,
    const unsigned char* __restrict__ pad) {
  __shared__ __attribute__((aligned(16))) unsigned char lds[32768 + 32768 + 16384];
  __bf16* sKV = (__bf16*)lds;             // 16384 elems: K half [256][64] or V^T half [64][256]
  float*  sP  = (float*)(lds + 32768);    // 16 x 512 fp32 scores
  __bf16* sPb = (__bf16*)(lds + 65536);   // 16 x 512 bf16 probabilities

  const int tid   = threadIdx.x;
  const int lane  = tid & 31;
  const int w     = tid >> 5;            // wave 0..3
  const int mlane = lane & 15;
  const int kh    = lane >> 4;
  const int qt = blockIdx.x;
  const int h  = blockIdx.y;
  const int b  = blockIdx.z;
  const size_t base = (size_t)b * S_ * E_;
  const int hc = h * HD_;
  const v8f vzero = {0.f, 0.f, 0.f, 0.f, 0.f, 0.f, 0.f, 0.f};

  // ---- preload Q fragments (one per 32-wide k chunk), reused for all kt ----
  v16bf aq[2];
#pragma unroll
  for (int kc = 0; kc < 2; ++kc) {
    v16bf a;
#pragma unroll
    for (int i = 0; i < 8; ++i) {
      int kb = i * 2 + 16 * kh;
      const float2 qv = *reinterpret_cast<const float2*>(
          &Q[base + (size_t)(qt * 16 + mlane) * E_ + hc + kc * 32 + kb]);
      a[2 * i]     = f2bf(qv.x);
      a[2 * i + 1] = f2bf(qv.y);
    }
    aq[kc] = a;
  }

  // ---- Phase 1: scores, two 256-key halves ----
  for (int half = 0; half < 2; ++half) {
    __syncthreads();
    // stage K half: sKV[key][d], keys half*256..+256, d = 0..63
    {
      const int krow = tid >> 3;        // 0..15 per iteration
      const int seg  = tid & 7;         // 8-float segment
#pragma unroll
      for (int j = 0; j < 16; ++j) {
        const int key = j * 16 + krow;
        const float* src =
            &Kt[base + (size_t)(half * 256 + key) * E_ + hc + seg * 8];
        const float4 f0 = *reinterpret_cast<const float4*>(src);
        const float4 f1 = *reinterpret_cast<const float4*>(src + 4);
        *reinterpret_cast<v8bf*>(&sKV[key * 64 + seg * 8]) = pack8(f0, f1);
      }
    }
    __syncthreads();
    // each wave computes 4 key tiles
#pragma unroll
    for (int j = 0; j < 4; ++j) {
      const int ktloc = j * 4 + w;        // 0..15 within half
      const int kt    = half * 16 + ktloc;
      v8f acc = vzero;
#pragma unroll
      for (int kc = 0; kc < 2; ++kc) {
        const int off = (ktloc * 16 + mlane) * 64 + kc * 32 + 16 * kh;
        const v8bf blo = *reinterpret_cast<const v8bf*>(&sKV[off]);
        const v8bf bhi = *reinterpret_cast<const v8bf*>(&sKV[off + 8]);
        acc = wmma_bf16(aq[kc], cat8(blo, bhi), acc);
      }
      // per-lane key column is invariant over i: one mask byte load per tile
      const int col = kt * 16 + mlane;
      bool pm = false;
      if (PAD) pm = (pad[b * S_ + col] != 0);
#pragma unroll
      for (int i = 0; i < 8; ++i) {
        const int m = i + 8 * kh;
        float s = acc[i] * 0.125f;                    // 1/sqrt(HD)
        bool masked = pm;
        if (CAUSAL) masked = masked || (col > qt * 16 + m);
        if (masked) s = NEG_;
        sP[m * S_ + col] = s;
      }
    }
  }
  __syncthreads();

  // ---- Phase 2: softmax, wave w handles rows 4w..4w+3; emit bf16 probs ----
#pragma unroll
  for (int r = 0; r < 4; ++r) {
    const int m = w * 4 + r;
    float mx = -3.4e38f;
    for (int c = lane; c < S_; c += 32) mx = fmaxf(mx, sP[m * S_ + c]);
#pragma unroll
    for (int o = 16; o > 0; o >>= 1) mx = fmaxf(mx, __shfl_xor(mx, o, 32));
    float sum = 0.f;
    for (int c = lane; c < S_; c += 32) {
      float e = __expf(sP[m * S_ + c] - mx);
      sP[m * S_ + c] = e;
      sum += e;
    }
#pragma unroll
    for (int o = 16; o > 0; o >>= 1) sum += __shfl_xor(sum, o, 32);
    const float inv = 1.f / sum;
    for (int c = lane; c < S_; c += 32)
      sPb[m * S_ + c] = f2bf(sP[m * S_ + c] * inv);
  }

  // ---- Phase 3: out = P(16x512) @ V(512x64); wave w -> columns [16w,16w+16) ----
  v8f accO = vzero;
  for (int half = 0; half < 2; ++half) {
    __syncthreads();   // sKV free; sPb complete
    // stage V^T half: sVT[d][key], d = tid&63, keys in 128-chunks
    {
      const int col = tid & 63;
      const int kb2 = (tid >> 6) * 128;
#pragma unroll
      for (int g = 0; g < 16; ++g) {
        v8bf p;
#pragma unroll
        for (int j = 0; j < 8; ++j) {
          const int key = half * 256 + kb2 + g * 8 + j;
          p[j] = f2bf(Vt[base + (size_t)key * E_ + hc + col]);
        }
        *reinterpret_cast<v8bf*>(&sKV[col * 256 + kb2 + g * 8]) = p;
      }
    }
    __syncthreads();
#pragma unroll
    for (int k0 = 0; k0 < 256; k0 += 32) {
      const int pa = mlane * S_ + half * 256 + k0;
      const v8bf alo = *reinterpret_cast<const v8bf*>(&sPb[pa + 16 * kh]);
      const v8bf ahi = *reinterpret_cast<const v8bf*>(&sPb[pa + 16 * kh + 8]);
      const int vb = (w * 16 + mlane) * 256 + k0 + 16 * kh;
      const v8bf blo = *reinterpret_cast<const v8bf*>(&sKV[vb]);
      const v8bf bhi = *reinterpret_cast<const v8bf*>(&sKV[vb + 8]);
      accO = wmma_bf16(cat8(alo, ahi), cat8(blo, bhi), accO);
    }
  }
#pragma unroll
  for (int i = 0; i < 8; ++i) {
    const int m = i + 8 * kh;
    O[base + (size_t)(qt * 16 + m) * E_ + hc + w * 16 + mlane] = accO[i];
  }
}

// ---------------------------------------------------------------------------
// x = LayerNorm(x + r) * g + b   (in-place on X; one wave per token row)
// ---------------------------------------------------------------------------
__global__ __launch_bounds__(32) void add_ln(
    float* __restrict__ X, const float* __restrict__ R,
    const float* __restrict__ g, const float* __restrict__ bta) {
  const int row  = blockIdx.x;
  const int lane = threadIdx.x;
  float v[E_ / 32];
  float sum = 0.f;
#pragma unroll
  for (int j = 0; j < E_ / 32; ++j) {
    int c = lane + j * 32;
    v[j] = X[(size_t)row * E_ + c] + R[(size_t)row * E_ + c];
    sum += v[j];
  }
#pragma unroll
  for (int o = 16; o > 0; o >>= 1) sum += __shfl_xor(sum, o, 32);
  float mean = sum * (1.0f / E_);
  float var = 0.f;
#pragma unroll
  for (int j = 0; j < E_ / 32; ++j) {
    float d = v[j] - mean;
    var += d * d;
  }
#pragma unroll
  for (int o = 16; o > 0; o >>= 1) var += __shfl_xor(var, o, 32);
  float rs = rsqrtf(var * (1.0f / E_) + EPS_);
#pragma unroll
  for (int j = 0; j < E_ / 32; ++j) {
    int c = lane + j * 32;
    X[(size_t)row * E_ + c] = (v[j] - mean) * rs * g[c] + bta[c];
  }
}

// ---------------------------------------------------------------------------
// Host: chain the full encoder-decoder forward pass on `stream`.
// ---------------------------------------------------------------------------
extern "C" void kernel_launch(void* const* d_in, const int* in_sizes, int n_in,
                              void* d_out, int out_size, void* d_ws,
                              size_t ws_size, hipStream_t stream) {
  (void)in_sizes; (void)n_in; (void)out_size; (void)ws_size;

  const int*           src_ids = (const int*)d_in[0];
  const int*           trg_ids = (const int*)d_in[1];
  const unsigned char* pad     = (const unsigned char*)d_in[2];
  const float* emb_src = (const float*)d_in[3];
  const float* emb_trg = (const float*)d_in[4];
  const float* enc_wq = (const float*)d_in[5];
  const float* enc_wk = (const float*)d_in[6];
  const float* enc_wv = (const float*)d_in[7];
  const float* enc_wo = (const float*)d_in[8];
  const float* enc_ln1_g = (const float*)d_in[9];
  const float* enc_ln1_b = (const float*)d_in[10];
  const float* enc_w1 = (const float*)d_in[11];
  const float* enc_b1 = (const float*)d_in[12];
  const float* enc_w2 = (const float*)d_in[13];
  const float* enc_b2 = (const float*)d_in[14];
  const float* enc_ln2_g = (const float*)d_in[15];
  const float* enc_ln2_b = (const float*)d_in[16];
  const float* dec_wq = (const float*)d_in[17];
  const float* dec_wk = (const float*)d_in[18];
  const float* dec_wv = (const float*)d_in[19];
  const float* dec_wo = (const float*)d_in[20];
  const float* dec_ln1_g = (const float*)d_in[21];
  const float* dec_ln1_b = (const float*)d_in[22];
  const float* dec_cwq = (const float*)d_in[23];
  const float* dec_cwk = (const float*)d_in[24];
  const float* dec_cwv = (const float*)d_in[25];
  const float* dec_cwo = (const float*)d_in[26];
  const float* dec_ln2_g = (const float*)d_in[27];
  const float* dec_ln2_b = (const float*)d_in[28];
  const float* dec_w1 = (const float*)d_in[29];
  const float* dec_b1 = (const float*)d_in[30];
  const float* dec_w2 = (const float*)d_in[31];
  const float* dec_b2 = (const float*)d_in[32];
  const float* dec_ln3_g = (const float*)d_in[33];
  const float* dec_ln3_b = (const float*)d_in[34];
  const float* w_out = (const float*)d_in[35];

  const int M = B_ * S_;                 // 2048 token rows
  const size_t XE = (size_t)M * E_;      // 1M floats per activation buffer

  float* ws  = (float*)d_ws;
  float* ENC = ws;                       // encoder stream / enc_out
  float* Y   = ws + 1 * XE;              // decoder stream
  float* Qb  = ws + 2 * XE;
  float* Kb  = ws + 3 * XE;
  float* Vb  = ws + 4 * XE;
  float* AO  = ws + 5 * XE;
  float* H1  = ws + 6 * XE;              // B*S*F = 4*XE
  float* Pb  = Qb;                       // alias: free after attention
  float* H2  = Kb;                       // alias: free after attention

  const dim3 blk(256);
  // projections: N=512, K=512, no bias
  auto proj = [&](const float* A, const float* W, float* C) {
    gemm_bf16_wmma<E_, E_, false, false>
        <<<dim3(E_ / 128, M / 64), blk, 0, stream>>>(A, W, nullptr, C);
  };
  // FFN up: N=2048, K=512, bias + relu
  auto ffn1 = [&](const float* A, const float* W, const float* bias, float* C) {
    gemm_bf16_wmma<F_, E_, true, true>
        <<<dim3(F_ / 128, M / 64), blk, 0, stream>>>(A, W, bias, C);
  };
  // FFN down: N=512, K=2048, bias
  auto ffn2 = [&](const float* A, const float* W, const float* bias, float* C) {
    gemm_bf16_wmma<E_, F_, false, true>
        <<<dim3(E_ / 128, M / 64), blk, 0, stream>>>(A, W, bias, C);
  };
  const dim3 gAttn(S_ / 16, H_, B_);
  const dim3 aBlk(128);

  // ---------------- encoder ----------------
  embed_pe<<<(M * E_) / 256, 256, 0, stream>>>(src_ids, emb_src, ENC);
  for (int l = 0; l < L_; ++l) {
    const size_t wEE = (size_t)l * E_ * E_;
    const size_t wEF = (size_t)l * E_ * F_;
    proj(ENC, enc_wq + wEE, Qb);
    proj(ENC, enc_wk + wEE, Kb);
    proj(ENC, enc_wv + wEE, Vb);
    attn_wmma<false, true><<<gAttn, aBlk, 0, stream>>>(Qb, Kb, Vb, AO, pad);
    proj(AO, enc_wo + wEE, Pb);
    add_ln<<<M, 32, 0, stream>>>(ENC, Pb, enc_ln1_g + l * E_, enc_ln1_b + l * E_);
    ffn1(ENC, enc_w1 + wEF, enc_b1 + (size_t)l * F_, H1);
    ffn2(H1, enc_w2 + wEF, enc_b2 + (size_t)l * E_, H2);
    add_ln<<<M, 32, 0, stream>>>(ENC, H2, enc_ln2_g + l * E_, enc_ln2_b + l * E_);
  }

  // ---------------- decoder ----------------
  embed_pe<<<(M * E_) / 256, 256, 0, stream>>>(trg_ids, emb_trg, Y);
  for (int l = 0; l < L_; ++l) {
    const size_t wEE = (size_t)l * E_ * E_;
    const size_t wEF = (size_t)l * E_ * F_;
    // causal self-attention
    proj(Y, dec_wq + wEE, Qb);
    proj(Y, dec_wk + wEE, Kb);
    proj(Y, dec_wv + wEE, Vb);
    attn_wmma<true, false><<<gAttn, aBlk, 0, stream>>>(Qb, Kb, Vb, AO, nullptr);
    proj(AO, dec_wo + wEE, Pb);
    add_ln<<<M, 32, 0, stream>>>(Y, Pb, dec_ln1_g + l * E_, dec_ln1_b + l * E_);
    // cross-attention (K,V from encoder output)
    proj(Y,   dec_cwq + wEE, Qb);
    proj(ENC, dec_cwk + wEE, Kb);
    proj(ENC, dec_cwv + wEE, Vb);
    attn_wmma<false, true><<<gAttn, aBlk, 0, stream>>>(Qb, Kb, Vb, AO, pad);
    proj(AO, dec_cwo + wEE, Pb);
    add_ln<<<M, 32, 0, stream>>>(Y, Pb, dec_ln2_g + l * E_, dec_ln2_b + l * E_);
    // FFN
    ffn1(Y, dec_w1 + wEF, dec_b1 + (size_t)l * F_, H1);
    ffn2(H1, dec_w2 + wEF, dec_b2 + (size_t)l * E_, H2);
    add_ln<<<M, 32, 0, stream>>>(Y, H2, dec_ln3_g + l * E_, dec_ln3_b + l * E_);
  }

  // ---------------- logits ----------------
  gemm_bf16_wmma<V_, E_, false, false>
      <<<dim3(V_ / 128, M / 64), blk, 0, stream>>>(Y, w_out, nullptr,
                                                   (float*)d_out);
}
